// SelfAttention_22342419874292
// MI455X (gfx1250) — matrix operand order repaired and checked by reading, not compiled, not run on hardware
//
#include <hip/hip_runtime.h>

// ---------------------------------------------------------------- constants
#define N_EMBD 1024
#define N_HEAD 16
#define HD     64
#define BATCH  8
#define SEQ    1024
#define M_TOT  (BATCH * SEQ)   // 8192 rows for all projections

typedef __bf16 bf16;
typedef __attribute__((ext_vector_type(16))) __bf16 bf16x16;
typedef __attribute__((ext_vector_type(8)))  float  floatx8;

// ---------------------------------------------------------------- WMMA helpers
// D = A(16x32 bf16) * B(32x16 bf16) + C(16x16 f32), wave32.
__device__ __forceinline__ floatx8 wmma_bf16(bf16x16 a, bf16x16 b, floatx8 c) {
    return __builtin_amdgcn_wmma_f32_16x16x32_bf16(
        /*neg_a=*/false, a, /*neg_b=*/false, b,
        /*c_mod=*/(short)0, c, /*reuse_a=*/false, /*reuse_b=*/false);
}

// A fragment (16x32, row-major source, row stride in elements).
// ISA layout: lanes 0-15 row M=lane hold K {0..7,16..23}; lanes 16-31 hold K {8..15,24..31}.
__device__ __forceinline__ bf16x16 load_A_frag(const bf16* p, int stride) {
    const int lane = threadIdx.x & 31;
    const bf16* r = p + (lane & 15) * stride;
    const int kb = (lane >> 4) << 3;          // 0 or 8
    bf16x16 a;
#pragma unroll
    for (int i = 0; i < 8; ++i) a[i] = r[kb + i];
#pragma unroll
    for (int i = 0; i < 8; ++i) a[8 + i] = r[16 + kb + i];
    return a;
}

// B fragment (32x16). Source is N x K row-major (column n of B == row n of source,
// contiguous over k). Lanes 0-15 (col n=lane) hold K {0..7,16..23}; lanes 16-31 hold K {8..15,24..31}.
__device__ __forceinline__ bf16x16 load_B_frag(const bf16* p, int stride) {
    const int lane = threadIdx.x & 31;
    const bf16* r = p + (lane & 15) * stride;
    const int kb = (lane >> 4) << 3;
    bf16x16 b;
#pragma unroll
    for (int i = 0; i < 8; ++i) b[i] = r[kb + i];
#pragma unroll
    for (int i = 0; i < 8; ++i) b[8 + i] = r[16 + kb + i];
    return b;
}

// ---------------------------------------------------------------- fp32 -> bf16 weights
__global__ __launch_bounds__(256)
void cvt_f32_to_bf16(const float* __restrict__ s, bf16* __restrict__ d, int n) {
    int i = (blockIdx.x * blockDim.x + threadIdx.x) * 4;
    if (i + 3 < n) {
        float4 f = *reinterpret_cast<const float4*>(s + i);
        d[i]     = (bf16)f.x;
        d[i + 1] = (bf16)f.y;
        d[i + 2] = (bf16)f.z;
        d[i + 3] = (bf16)f.w;
    }
}

// ---------------------------------------------------------------- WMMA GEMM
// out[m][n] = sum_k X[m][k] * W[n][k] + bias[n]   (torch Linear: x @ W.T + b)
// X: fp32 (M_TOT x N_EMBD), converted to bf16 while staging into LDS.
// W: bf16 (N_EMBD x N_EMBD) row-major (row n contiguous over k) -> B fragments directly.
// Workgroup tile 128x128: 8 waves as 4 (M) x 2 (N); each wave owns 32x64.
// K-depth 64 per stage -> 16 WMMAs per barrier pair (2 A-frags x 4 B-frags x 2 k-steps).
// MODE 0: fp32 out[m*N_EMBD+n]                      (final projection)
// MODE 1: bf16 head layout  out[(bh*SEQ+l)*HD+dh]   (Q with scale=0.125, K)
// MODE 2: bf16 transposed   out[(bh*HD+dh)*SEQ+l]   (V^T for the P*V WMMA)
#define KT    64
#define MT    128
#define NTILE 128
#define ASTR  72   // padded LDS row stride (elements): 36 dwords/row -> 16 rows on 16 banks

template <int MODE>
__global__ __launch_bounds__(256)
void gemm_wmma(const float* __restrict__ X, const bf16* __restrict__ W,
               const float* __restrict__ bias, void* __restrict__ out,
               float scale) {
    __shared__ bf16 aT[MT * ASTR];      // 18432 B
    __shared__ bf16 bT[NTILE * ASTR];   // 18432 B

    const int tid  = threadIdx.x;
    const int wave = tid >> 5;
    const int lane = tid & 31;
    const int wm   = wave >> 1;         // 0..3  -> 32-row slice
    const int wn   = wave & 1;          // 0..1  -> 64-col slice
    const int m0   = blockIdx.y * MT;
    const int n0   = blockIdx.x * NTILE;

    floatx8 acc0[4], acc1[4];
#pragma unroll
    for (int c = 0; c < 4; ++c) {
        acc0[c] = floatx8{0.f,0.f,0.f,0.f,0.f,0.f,0.f,0.f};
        acc1[c] = floatx8{0.f,0.f,0.f,0.f,0.f,0.f,0.f,0.f};
    }

    const int srow = tid >> 1;          // 0..127, 32 elements (half a K-slab) each
    const int scb  = (tid & 1) * 32;

    for (int k0 = 0; k0 < N_EMBD; k0 += KT) {
        // stage A tile: fp32 -> bf16 into LDS (128 x 64)
        {
            const float* src = X + (size_t)(m0 + srow) * N_EMBD + k0 + scb;
            __builtin_prefetch(src + KT, 0, 0);           // global_prefetch_b8 next tile
            bf16* dst = &aT[srow * ASTR + scb];
#pragma unroll
            for (int q = 0; q < 8; ++q) {
                float4 f = reinterpret_cast<const float4*>(src)[q];
                dst[4 * q]     = (bf16)f.x;
                dst[4 * q + 1] = (bf16)f.y;
                dst[4 * q + 2] = (bf16)f.z;
                dst[4 * q + 3] = (bf16)f.w;
            }
        }
        // stage B tile: bf16 copy (128 x 64)
        {
            const bf16* src = W + (size_t)(n0 + srow) * N_EMBD + k0 + scb;
            bf16* dst = &bT[srow * ASTR + scb];
#pragma unroll
            for (int i = 0; i < 32; ++i) dst[i] = src[i];
        }
        __syncthreads();

#pragma unroll
        for (int ks = 0; ks < KT; ks += 32) {
            bf16x16 af0 = load_A_frag(&aT[(wm * 32)      * ASTR + ks], ASTR);
            bf16x16 af1 = load_A_frag(&aT[(wm * 32 + 16) * ASTR + ks], ASTR);
#pragma unroll
            for (int c = 0; c < 4; ++c) {
                bf16x16 bfc = load_B_frag(&bT[(wn * 64 + c * 16) * ASTR + ks], ASTR);
                acc0[c] = wmma_bf16(af0, bfc, acc0[c]);
                acc1[c] = wmma_bf16(af1, bfc, acc1[c]);
            }
        }
        __syncthreads();
    }

    // epilogue from D layout: lanes 0-15 -> col n=lane, rows = vgpr 0..7;
    // lanes 16-31 -> col n=lane-16, rows = vgpr+8.
    const int mbase = (lane >> 4) << 3;
    const int nlane = lane & 15;
#pragma unroll
    for (int c = 0; c < 4; ++c) {
        const int n = n0 + wn * 64 + c * 16 + nlane;
        const float bval = bias[n];
#pragma unroll
        for (int half = 0; half < 2; ++half) {
#pragma unroll
            for (int r = 0; r < 8; ++r) {
                const int m = m0 + wm * 32 + half * 16 + mbase + r;
                const float v = ((half ? acc1[c][r] : acc0[c][r]) + bval) * scale;
                if (MODE == 0) {
                    ((float*)out)[(size_t)m * N_EMBD + n] = v;
                } else {
                    const int b = m >> 10, l = m & 1023, h = n >> 6, dh = n & 63;
                    const int bh = b * N_HEAD + h;
                    if (MODE == 1)
                        ((bf16*)out)[((size_t)bh * SEQ + l) * HD + dh] = (bf16)v;
                    else
                        ((bf16*)out)[((size_t)bh * HD + dh) * SEQ + l] = (bf16)v;
                }
            }
        }
    }
}

// ---------------------------------------------------------------- flash attention
// One wave per 16 query rows; 8 waves (256 thr) per block cover 128 rows of one (b,h).
// Per 32-key block: S = Q*K^T (4 WMMAs), causal mask + online softmax in D-layout
// registers (row stats reduced over the 16-lane half via shfl_xor), P transposed
// through per-wave LDS into an A fragment, O += P*V^T (4 WMMAs).
#define PSTR 36

__global__ __launch_bounds__(256)
void attn_wmma(const bf16* __restrict__ Q, const bf16* __restrict__ Kh,
               const bf16* __restrict__ VT, float* __restrict__ Y) {
    __shared__ bf16 pbuf[8][16 * PSTR];   // per-wave 16x32 P tile, padded

    const int wave = threadIdx.x >> 5;
    const int lane = threadIdx.x & 31;
    const int bh   = blockIdx.x >> 3;
    const int q0   = (blockIdx.x & 7) * 128 + wave * 16;
    const int b    = bh >> 4, h = bh & 15;

    const bf16* Qp = Q  + ((size_t)bh * SEQ + q0) * HD;
    const bf16* Kp = Kh + (size_t)bh * SEQ * HD;
    const bf16* Vp = VT + (size_t)bh * HD * SEQ;

    const bf16x16 qa0 = load_A_frag(Qp, HD);        // hd 0..31  (Q pre-scaled by 1/8)
    const bf16x16 qa1 = load_A_frag(Qp + 32, HD);   // hd 32..63

    float mrow[8], lrow[8];
    floatx8 o[4];
#pragma unroll
    for (int r = 0; r < 8; ++r) { mrow[r] = -__builtin_inff(); lrow[r] = 0.f; }
#pragma unroll
    for (int c = 0; c < 4; ++c) o[c] = floatx8{0.f,0.f,0.f,0.f,0.f,0.f,0.f,0.f};

    const int mbase = (lane >> 4) << 3;
    const int nlane = lane & 15;
    const int kbmax = (q0 + 15) >> 5;               // wave-local causal bound

    for (int kb = 0; kb <= kbmax; ++kb) {
        const int key0 = kb * 32;
        const floatx8 z = floatx8{0.f,0.f,0.f,0.f,0.f,0.f,0.f,0.f};

        // S tiles: cols = key positions (K rows contiguous over hd -> fast B frags)
        bf16x16 b00 = load_B_frag(Kp + (size_t)key0 * HD,             HD);
        bf16x16 b01 = load_B_frag(Kp + (size_t)key0 * HD + 32,        HD);
        floatx8 s0 = wmma_bf16(qa0, b00, z); s0 = wmma_bf16(qa1, b01, s0);
        bf16x16 b10 = load_B_frag(Kp + (size_t)(key0 + 16) * HD,      HD);
        bf16x16 b11 = load_B_frag(Kp + (size_t)(key0 + 16) * HD + 32, HD);
        floatx8 s1 = wmma_bf16(qa0, b10, z); s1 = wmma_bf16(qa1, b11, s1);

        // causal mask + online softmax (row r / r+8 lives in vgpr slot r per lane half)
#pragma unroll
        for (int r = 0; r < 8; ++r) {
            const int row = q0 + mbase + r;
            float v0 = (key0 + nlane      <= row) ? s0[r] : -__builtin_inff();
            float v1 = (key0 + 16 + nlane <= row) ? s1[r] : -__builtin_inff();
            float mx = fmaxf(v0, v1);
#pragma unroll
            for (int d = 1; d < 16; d <<= 1) mx = fmaxf(mx, __shfl_xor(mx, d, 32));
            const float mnew = fmaxf(mrow[r], mx);
            const float corr = __expf(mrow[r] - mnew);   // first block: exp(-inf)=0, l=o=0
            mrow[r] = mnew;
            const float p0 = __expf(v0 - mnew);
            const float p1 = __expf(v1 - mnew);
            float rs = p0 + p1;
#pragma unroll
            for (int d = 1; d < 16; d <<= 1) rs += __shfl_xor(rs, d, 32);
            lrow[r] = lrow[r] * corr + rs;
#pragma unroll
            for (int c = 0; c < 4; ++c) o[c][r] *= corr;
            pbuf[wave][(mbase + r) * PSTR + nlane]      = (bf16)p0;
            pbuf[wave][(mbase + r) * PSTR + 16 + nlane] = (bf16)p1;
        }

        // wave-local LDS store->load ordering (no block barrier: bounds are non-uniform)
        asm volatile("s_wait_dscnt 0x0" ::: "memory");

        bf16x16 pa = load_A_frag(&pbuf[wave][0], PSTR);
#pragma unroll
        for (int c = 0; c < 4; ++c) {
            bf16x16 bv = load_B_frag(Vp + (size_t)(c * 16) * SEQ + key0, SEQ);
            o[c] = wmma_bf16(pa, bv, o[c]);
        }
    }

    // normalize and write Y back to (b, l, h*64+dh) fp32 for the output projection
#pragma unroll
    for (int r = 0; r < 8; ++r) {
        const int row = q0 + mbase + r;
        const float inv = 1.f / lrow[r];
#pragma unroll
        for (int c = 0; c < 4; ++c) {
            Y[((size_t)b * SEQ + row) * N_EMBD + h * HD + c * 16 + nlane] = o[c][r] * inv;
        }
    }
}

// ---------------------------------------------------------------- host launch
extern "C" void kernel_launch(void* const* d_in, const int* in_sizes, int n_in,
                              void* d_out, int out_size, void* d_ws, size_t ws_size,
                              hipStream_t stream) {
    (void)in_sizes; (void)n_in; (void)out_size; (void)ws_size;

    const float* x_key = (const float*)d_in[0];
    const float* x_val = (const float*)d_in[1];
    const float* x_qry = (const float*)d_in[2];
    const float* Wk = (const float*)d_in[3];
    const float* bk = (const float*)d_in[4];
    const float* Wq = (const float*)d_in[5];
    const float* bq = (const float*)d_in[6];
    const float* Wv = (const float*)d_in[7];
    const float* bv = (const float*)d_in[8];
    const float* Wp = (const float*)d_in[9];
    const float* bp = (const float*)d_in[10];

    // workspace carve-up (~92 MB total)
    char* ws = (char*)d_ws;
    size_t off = 0;
    auto carve = [&](size_t bytes) {
        void* p = ws + off;
        off += (bytes + 255) & ~(size_t)255;
        return p;
    };
    const size_t SZ_W = (size_t)N_EMBD * N_EMBD * sizeof(bf16);           // 2 MB
    const size_t SZ_H = (size_t)BATCH * N_HEAD * SEQ * HD * sizeof(bf16); // 16 MB
    const size_t SZ_Y = (size_t)M_TOT * N_EMBD * sizeof(float);           // 32 MB

    bf16*  wk_bf = (bf16*)carve(SZ_W);
    bf16*  wq_bf = (bf16*)carve(SZ_W);
    bf16*  wv_bf = (bf16*)carve(SZ_W);
    bf16*  wp_bf = (bf16*)carve(SZ_W);
    bf16*  Qh    = (bf16*)carve(SZ_H);
    bf16*  Kh    = (bf16*)carve(SZ_H);
    bf16*  VT    = (bf16*)carve(SZ_H);
    float* Ybuf  = (float*)carve(SZ_Y);

    // 1) weights fp32 -> bf16
    const int nW = N_EMBD * N_EMBD;
    const int cvtGrid = nW / (256 * 4);
    cvt_f32_to_bf16<<<cvtGrid, 256, 0, stream>>>(Wk, wk_bf, nW);
    cvt_f32_to_bf16<<<cvtGrid, 256, 0, stream>>>(Wq, wq_bf, nW);
    cvt_f32_to_bf16<<<cvtGrid, 256, 0, stream>>>(Wv, wv_bf, nW);
    cvt_f32_to_bf16<<<cvtGrid, 256, 0, stream>>>(Wp, wp_bf, nW);

    // 2) QKV projections (Q carries the 1/sqrt(64) softmax scale)
    dim3 gGrid(N_EMBD / NTILE, M_TOT / MT);  // (8, 64)
    gemm_wmma<1><<<gGrid, 256, 0, stream>>>(x_key, wk_bf, bk, Kh, 1.0f);
    gemm_wmma<1><<<gGrid, 256, 0, stream>>>(x_qry, wq_bf, bq, Qh, 0.125f);
    gemm_wmma<2><<<gGrid, 256, 0, stream>>>(x_val, wv_bf, bv, VT, 1.0f);

    // 3) causal flash attention: 8*16 heads * 8 row-blocks = 1024 workgroups
    attn_wmma<<<BATCH * N_HEAD * (SEQ / 128), 256, 0, stream>>>(Qh, Kh, VT, Ybuf);

    // 4) output projection -> fp32 d_out
    gemm_wmma<0><<<gGrid, 256, 0, stream>>>(Ybuf, wp_bf, bp, (float*)d_out, 1.0f);
}
